// ImportanceRenderer_77670188581460
// MI455X (gfx1250) — matrix operand order repaired and checked by reading, not compiled, not run on hardware
//
#include <hip/hip_runtime.h>
#include <hip/hip_bf16.h>
#include <math.h>

// ---------------------------------------------------------------------------
// ImportanceRenderer forward for MI455X (gfx1250, wave32).
//
// Perf model: triplane gather (2.4 GB scattered, L2-resident: planes = 100.6MB
// < 192MB L2) dominates; MLP (13 GFLOP) mapped to v_wmma_f32_16x16x32_f16
// (16 samples / wave tile); per-ray stages are trivial scalar work.
// ---------------------------------------------------------------------------

typedef __attribute__((ext_vector_type(16))) _Float16 v16h;
typedef __attribute__((ext_vector_type(8)))  float    v8f;

#define DRES  48      // depth_resolution == depth_resolution_importance (setup)
#define NCH   32      // plane channels
#define HID   64      // hidden width
#define NOUT  33      // MLP outputs (1 sigma + 32 rgb)
#define NOUTP 48      // padded to 3 WMMA N-tiles
#define HRES  256
#define WRES  256

__device__ __forceinline__ float softplus_f(float x) {
  // logaddexp(x, 0) — overflow-safe, matches jax.nn.softplus
  return fmaxf(x, 0.0f) + log1pf(expf(-fabsf(x)));
}
__device__ __forceinline__ float sigmoid_f(float x) {
  return 1.0f / (1.0f + expf(-x));
}

// t_min/t_max per batch: obs_bbox_cam z-range with near/far fallback
__device__ __forceinline__ void batch_tminmax(const float* __restrict__ ro,
                                              const float* __restrict__ obs,
                                              int b, int R, float& tmin, float& tmax) {
  const float* o = ro + (size_t)b * R * 3;           // ray 0 of batch b
  float cd = sqrtf(o[0]*o[0] + o[1]*o[1] + o[2]*o[2]);
  float nearv = cd - 0.75f, farv = cd + 0.75f;
  float t0 = obs[b*6 + 2];                           // obs[b,0,2]
  float t1 = obs[b*6 + 3 + 2];                       // obs[b,1,2]
  bool bad = t0 >= t1;
  tmin = bad ? nearv : t0;
  tmax = bad ? farv  : t1;
}

// ---------------------------------------------------------------------------
// Kernel: fused triplane sample + MLP decode.
// Block = 128 threads = 4 waves; each wave owns 16 samples (one WMMA M-tile).
// Gather phase: 2 lanes per sample, 16 channels per lane (3 planes×4 corners).
// MLP phase: layer1 = 4 × wmma(16x16x32 f16), layer2 = 2 K-chunks × 3 N-tiles.
// ---------------------------------------------------------------------------
__global__ __launch_bounds__(128)
void k_sample_decode(const float* __restrict__ planes,
                     const float* __restrict__ w1, const float* __restrict__ b1,
                     const float* __restrict__ w2, const float* __restrict__ b2,
                     const float* __restrict__ ro, const float* __restrict__ rd,
                     const float* __restrict__ canon, const float* __restrict__ obs,
                     const float* __restrict__ zfine,
                     float* __restrict__ rgb_out, float* __restrict__ sig_out,
                     int B, int R, int mode)
{
  __shared__ float sW1[NCH * HID];
  __shared__ float sB1[HID];
  __shared__ float sW2[HID * NOUTP];
  __shared__ float sB2[NOUTP];
  __shared__ float sFeat[4][16][NCH];
  __shared__ float sHid[4][16][HID];

  const int tid = threadIdx.x;

  // Stage MLP weights in LDS (w2 columns padded 33 -> 48 with zeros).
  for (int i = tid; i < NCH * HID; i += 128) sW1[i] = w1[i];
  for (int i = tid; i < HID; i += 128)       sB1[i] = b1[i];
  for (int i = tid; i < HID * NOUTP; i += 128) {
    int k = i / NOUTP, n = i % NOUTP;
    sW2[i] = (n < NOUT) ? w2[k * NOUT + n] : 0.0f;
  }
  for (int i = tid; i < NOUTP; i += 128) sB2[i] = (i < NOUT) ? b2[i] : 0.0f;

  const int wave = tid >> 5;
  const int lane = tid & 31;
  const int half = lane >> 4;     // which 16-lane half of the wave
  const int s    = lane & 15;     // sample slot / WMMA column

  const long long sampleBase = (long long)blockIdx.x * 64 + (long long)wave * 16;
  const long long gs = sampleBase + s;
  const int didx = (int)(gs % DRES);
  const long long ray = gs / DRES;
  const int r = (int)(ray % R);
  const int b = (int)(ray / R);

  float tmin, tmax;
  batch_tminmax(ro, obs, b, R, tmin, tmax);
  float z;
  if (mode == 0) z = tmin + (tmax - tmin) * ((float)didx / (float)(DRES - 1));
  else           z = zfine[gs];

  const float* o  = ro + ((size_t)b * R + r) * 3;
  const float* dv = rd + ((size_t)b * R + r) * 3;
  float cx = o[0] + z * dv[0];
  float cy = o[1] + z * dv[1];
  float cz = o[2] + z * dv[2];
  // normalize into [-1,1] by canonical bbox
  float nx = 2.0f * (cx - canon[b*6+0]) / (canon[b*6+3+0] - canon[b*6+0]) - 1.0f;
  float ny = 2.0f * (cy - canon[b*6+1]) / (canon[b*6+3+1] - canon[b*6+1]) - 1.0f;
  float nz = 2.0f * (cz - canon[b*6+2]) / (canon[b*6+3+2] - canon[b*6+2]) - 1.0f;

  // INV_PLANES projections: p0:(x,y)  p1:(z,x)  p2:(y,z)
  float gxs[3] = {nx, nz, ny};
  float gys[3] = {ny, nx, nz};

  float acc[16];
  #pragma unroll
  for (int c = 0; c < 16; ++c) acc[c] = 0.0f;
  const int ch0 = half * 16;

  #pragma unroll
  for (int p = 0; p < 3; ++p) {
    float x = (gxs[p] + 1.0f) * (0.5f * WRES) - 0.5f;
    float y = (gys[p] + 1.0f) * (0.5f * HRES) - 0.5f;
    float x0f = floorf(x), y0f = floorf(y);
    float wx1 = x - x0f, wy1 = y - y0f;
    int ix0 = (int)x0f, iy0 = (int)y0f;
    int   off[4];
    float wgt[4];
    #pragma unroll
    for (int kk = 0; kk < 4; ++kk) {
      int dx = kk & 1, dy = kk >> 1;
      int ix = ix0 + dx, iy = iy0 + dy;
      bool valid = (ix >= 0) && (ix < WRES) && (iy >= 0) && (iy < HRES);
      int cix = min(max(ix, 0), WRES - 1);
      int ciy = min(max(iy, 0), HRES - 1);
      off[kk] = ciy * WRES + cix;
      float wx = dx ? wx1 : 1.0f - wx1;
      float wy = dy ? wy1 : 1.0f - wy1;
      wgt[kk] = valid ? wx * wy : 0.0f;
    }
    const float* pbase = planes + (((size_t)b * 3 + p) * NCH + ch0) * (size_t)(HRES * WRES);
    __builtin_prefetch(pbase + off[0], 0, 1);   // global_prefetch_b8 hint into L2
    for (int c = 0; c < 16; ++c) {
      const float* cb = pbase + (size_t)c * (HRES * WRES);
      acc[c] += wgt[0]*cb[off[0]] + wgt[1]*cb[off[1]]
              + wgt[2]*cb[off[2]] + wgt[3]*cb[off[3]];
    }
  }
  #pragma unroll
  for (int c = 0; c < 16; ++c)
    sFeat[wave][s][ch0 + c] = acc[c] * (1.0f / 3.0f);   // mean over 3 planes

  __syncthreads();

  // ---- Layer 1: D(16x16) = A(16samples x 32feat, f16) x B(32 x 16hid) ----
  // A layout (16-bit A 16x32): lane m holds K = e + half*16 (matches B)
  v16h a;
  #pragma unroll
  for (int e = 0; e < 16; ++e) {
    int k = e + (half ? 16 : 0);
    a[e] = (_Float16)sFeat[wave][s][k];
  }
  #pragma unroll
  for (int t = 0; t < 4; ++t) {
    v16h bf;
    #pragma unroll
    for (int e = 0; e < 16; ++e) {
      int k = e + (half ? 16 : 0);              // B: lanes0-15 K=0..15, lanes16-31 K=16..31
      bf[e] = (_Float16)sW1[k * HID + (t * 16 + s)];
    }
    v8f c = {};
    c = __builtin_amdgcn_wmma_f32_16x16x32_f16(false, a, false, bf, (short)0, c, false, false);
    // D: lane holds rows M = j + half*8, column N = s
    #pragma unroll
    for (int j = 0; j < 8; ++j) {
      int m = j + half * 8;
      int h = t * 16 + s;
      sHid[wave][m][h] = softplus_f(c[j] + sB1[h]);
    }
  }
  __syncthreads();

  // ---- Layer 2: D(16x48) = H(16x64, f16) x W2(64x48), K split 2x32 ----
  v8f zero = {};
  v8f co[3] = {zero, zero, zero};
  #pragma unroll
  for (int cc = 0; cc < 2; ++cc) {
    v16h a2;
    #pragma unroll
    for (int e = 0; e < 16; ++e) {
      int k = 32 * cc + e + (half ? 16 : 0);
      a2[e] = (_Float16)sHid[wave][s][k];
    }
    #pragma unroll
    for (int t = 0; t < 3; ++t) {
      v16h bf;
      #pragma unroll
      for (int e = 0; e < 16; ++e) {
        int k = 32 * cc + e + (half ? 16 : 0);
        bf[e] = (_Float16)sW2[(k) * NOUTP + (t * 16 + s)];
      }
      co[t] = __builtin_amdgcn_wmma_f32_16x16x32_f16(false, a2, false, bf, (short)0,
                                                     co[t], false, false);
    }
  }

  // Epilogue: n==0 -> sigma; n in [1,33) -> rgb = sigmoid*1.002-0.001
  #pragma unroll
  for (int t = 0; t < 3; ++t) {
    int n = t * 16 + s;
    float bias = sB2[n];
    #pragma unroll
    for (int j = 0; j < 8; ++j) {
      int m = j + half * 8;
      long long gsm = sampleBase + m;
      float ov = co[t][j] + bias;
      if (n == 0) {
        sig_out[gsm] = ov;
      } else if (n < NOUT) {
        rgb_out[gsm * 32 + (n - 1)] = sigmoid_f(ov) * 1.002f - 0.001f;
      }
    }
  }
}

// ---------------------------------------------------------------------------
// Kernel: coarse ray-march + deterministic importance resampling.
// One thread per ray; u is monotone so searchsorted becomes a running pointer.
// ---------------------------------------------------------------------------
__global__ __launch_bounds__(256)
void k_march_importance(const float* __restrict__ sig_c,
                        const float* __restrict__ ro, const float* __restrict__ obs,
                        float* __restrict__ zfine, int B, int R)
{
  long long ray = (long long)blockIdx.x * 256 + threadIdx.x;
  long long NR = (long long)B * R;
  if (ray >= NR) return;
  int b = (int)(ray / R);

  float tmin, tmax;
  batch_tminmax(ro, obs, b, R, tmin, tmax);
  float dz = (tmax - tmin) / (float)(DRES - 1);

  float sig[DRES];
  for (int i = 0; i < DRES; ++i) sig[i] = sig_c[ray * DRES + i];

  // weights from coarse marching (delta is uniform)
  float wgt[DRES - 1];
  float T = 1.0f;
  for (int i = 0; i < DRES - 1; ++i) {
    float dmid  = 0.5f * (sig[i] + sig[i + 1]);
    float dens  = softplus_f(dmid - 1.0f);
    float alpha = 1.0f - expf(-dens * dz);
    wgt[i] = alpha * T;
    T *= (1.0f - alpha + 1e-10f);
  }

  // max-filter (pad -inf) then 2-tap smooth + 0.01
  float m[DRES];
  m[0] = wgt[0];
  for (int i = 1; i <= DRES - 2; ++i) m[i] = fmaxf(wgt[i - 1], wgt[i]);
  m[DRES - 1] = wgt[DRES - 2];
  float wsm[DRES - 1];
  for (int i = 0; i < DRES - 1; ++i) wsm[i] = 0.5f * (m[i] + m[i + 1]) + 0.01f;

  // CDF over interior weights wsm[1..45] (+eps), bins = zmid
  const int NSB = DRES - 3;          // 45
  float cdf[DRES - 2];               // 46
  float tot = 0.0f;
  for (int i = 0; i < NSB; ++i) tot += (wsm[i + 1] + 1e-5f);
  cdf[0] = 0.0f;
  float run = 0.0f;
  for (int i = 0; i < NSB; ++i) { run += (wsm[i + 1] + 1e-5f); cdf[i + 1] = run / tot; }

  // invert CDF at u = linspace(0,1,48), side='right'
  int k = 0;
  for (int j = 0; j < DRES; ++j) {
    float u = (float)j / (float)(DRES - 1);
    while (k < DRES - 2 && cdf[k] <= u) ++k;
    int below = k - 1; below = max(0, min(below, NSB));
    int above = min(k, NSB);
    float cb = cdf[below], ca = cdf[above];
    float bb = tmin + ((float)below + 0.5f) * dz;   // zmid bins
    float ba = tmin + ((float)above + 0.5f) * dz;
    float denom = (ca - cb < 1e-5f) ? 1.0f : (ca - cb);
    zfine[ray * DRES + j] = bb + (u - cb) / denom * (ba - bb);
  }
}

// ---------------------------------------------------------------------------
// Kernel: merge sorted coarse+fine depths (stable: coarse wins ties, matching
// stable argsort of concatenation) + final volumetric composite.
// One wave per ray; lane = rgb channel; merge control is wave-uniform.
// ---------------------------------------------------------------------------
__global__ __launch_bounds__(128)
void k_final(const float* __restrict__ rgb_c, const float* __restrict__ sig_c,
             const float* __restrict__ rgb_f, const float* __restrict__ sig_f,
             const float* __restrict__ zf,
             const float* __restrict__ ro, const float* __restrict__ obs,
             float* __restrict__ out, int B, int R)
{
  const int wave = threadIdx.x >> 5;
  const int lane = threadIdx.x & 31;
  const long long NR = (long long)B * R;
  const long long ray = (long long)blockIdx.x * 4 + wave;
  if (ray >= NR) return;
  const int b = (int)(ray / R);

  // global depth clamp = min/max over all batches' sample ranges
  float gmin = 1e30f, gmax = -1e30f;
  for (int bb = 0; bb < B; ++bb) {
    float t0, t1; batch_tminmax(ro, obs, bb, R, t0, t1);
    gmin = fminf(gmin, t0); gmax = fmaxf(gmax, t1);
  }
  float tmin, tmax;
  batch_tminmax(ro, obs, b, R, tmin, tmax);
  float dz = (tmax - tmin) / (float)(DRES - 1);

  const long long base = ray * DRES;
  int i = 0, j = 0;
  float T = 1.0f, accC = 0.0f, accW = 0.0f, accWZ = 0.0f;
  float pz = 0.0f, psig = 0.0f, prgb = 0.0f;
  for (int kk = 0; kk < 2 * DRES; ++kk) {
    float zc  = tmin + (float)i * dz;
    float zfv = (j < DRES) ? zf[base + j] : 0.0f;
    bool takeC = (i < DRES) && ((j >= DRES) || (zc <= zfv));
    float z, sg, rc;
    if (takeC) { z = zc;  sg = sig_c[base + i]; rc = rgb_c[(base + i) * 32 + lane]; ++i; }
    else       { z = zfv; sg = sig_f[base + j]; rc = rgb_f[(base + j) * 32 + lane]; ++j; }
    if (kk > 0) {
      float delta = z - pz;
      float dmid  = 0.5f * (sg + psig);
      float dens  = softplus_f(dmid - 1.0f);
      float alpha = 1.0f - expf(-dens * delta);
      float w = alpha * T;
      T *= (1.0f - alpha + 1e-10f);
      accC  += w * 0.5f * (rc + prgb);
      accW  += w;
      accWZ += w * 0.5f * (z + pz);
    }
    pz = z; psig = sg; prgb = rc;
  }
  out[ray * 32 + lane] = accC * 2.0f - 1.0f;
  if (lane == 0) {
    float depth = accWZ / accW;
    if (depth != depth) depth = 0.0f;               // nan_to_num
    depth = fminf(fmaxf(depth, gmin), gmax);
    out[NR * 32 + ray] = depth;
    out[NR * 33 + ray] = accW;
  }
}

// ---------------------------------------------------------------------------
extern "C" void kernel_launch(void* const* d_in, const int* in_sizes, int n_in,
                              void* d_out, int out_size, void* d_ws, size_t ws_size,
                              hipStream_t stream)
{
  const float* planes = (const float*)d_in[0];
  const float* w1     = (const float*)d_in[1];
  const float* b1     = (const float*)d_in[2];
  const float* w2     = (const float*)d_in[3];
  const float* b2     = (const float*)d_in[4];
  const float* ro     = (const float*)d_in[5];
  const float* rd     = (const float*)d_in[6];
  const float* canon  = (const float*)d_in[7];
  const float* obs    = (const float*)d_in[8];

  const int B = in_sizes[7] / 6;                 // canonical_bbox: B*2*3
  const int R = in_sizes[5] / (3 * B);           // ray_origins:    B*R*3
  const long long NS = (long long)B * R * DRES;  // samples per pass
  const long long NR = (long long)B * R;

  // workspace layout (floats): rgb_c | sig_c | rgb_f | sig_f | zfine  (~211MB)
  float* ws    = (float*)d_ws;
  float* rgb_c = ws;
  float* sig_c = rgb_c + NS * 32;
  float* rgb_f = sig_c + NS;
  float* sig_f = rgb_f + NS * 32;
  float* zfine = sig_f + NS;

  dim3 blk(128);
  dim3 grid_s((unsigned)((NS + 63) / 64));       // 64 samples (4 WMMA tiles) / block

  k_sample_decode<<<grid_s, blk, 0, stream>>>(planes, w1, b1, w2, b2, ro, rd, canon, obs,
                                              zfine, rgb_c, sig_c, B, R, /*mode=*/0);
  k_march_importance<<<dim3((unsigned)((NR + 255) / 256)), dim3(256), 0, stream>>>(
      sig_c, ro, obs, zfine, B, R);
  k_sample_decode<<<grid_s, blk, 0, stream>>>(planes, w1, b1, w2, b2, ro, rd, canon, obs,
                                              zfine, rgb_f, sig_f, B, R, /*mode=*/1);
  k_final<<<dim3((unsigned)((NR + 3) / 4)), blk, 0, stream>>>(
      rgb_c, sig_c, rgb_f, sig_f, zfine, ro, obs, (float*)d_out, B, R);
}
